// TopKSAE_41300405518696
// MI455X (gfx1250) — matrix operand order repaired and checked by reading, not compiled, not run on hardware
//
#include <hip/hip_runtime.h>

#define D_ACT   2048
#define N_FEAT  32768
#define SPARSITY 64
#define BATCH   8192
#define EPS     1e-6f

typedef __attribute__((ext_vector_type(16))) __bf16 v16bf;
typedef __attribute__((ext_vector_type(8)))  __bf16 v8bf;
typedef __attribute__((ext_vector_type(8)))  float  v8f;

// ---------- helpers ----------
__device__ __forceinline__ unsigned int pack_bf16x2(float lo, float hi) {
    unsigned int ulo = __float_as_uint(lo); ulo += 0x7FFFu + ((ulo >> 16) & 1u);
    unsigned int uhi = __float_as_uint(hi); uhi += 0x7FFFu + ((uhi >> 16) & 1u);
    return (ulo >> 16) | (uhi & 0xFFFF0000u);
}
// monotone map: larger float -> larger unsigned key
__device__ __forceinline__ unsigned int float_key(float v) {
    unsigned int u = __float_as_uint(v);
    return (u & 0x80000000u) ? ~u : (u | 0x80000000u);
}

// ---------- 1) inverse row norms of dict_w ----------
__global__ __launch_bounds__(256) void norm_kernel(const float* __restrict__ W,
                                                   float* __restrict__ inv_norm) {
    const int wave = threadIdx.x >> 5;
    const int lane = threadIdx.x & 31;
    const int row  = blockIdx.x * 8 + wave;
    const float4* p = reinterpret_cast<const float4*>(W + (size_t)row * D_ACT);
    float ss = 0.0f;
#pragma unroll 4
    for (int i = lane; i < D_ACT / 4; i += 32) {
        float4 v = p[i];
        ss += v.x * v.x + v.y * v.y + v.z * v.z + v.w * v.w;
    }
#pragma unroll
    for (int off = 16; off > 0; off >>= 1) ss += __shfl_xor(ss, off, 32);
    if (lane == 0) inv_norm[row] = 1.0f / (sqrtf(ss) + EPS);
}

// ---------- 2) W -> bf16, pre-scaled by inv_norm (folds normalization) ----------
__global__ __launch_bounds__(256) void conv_w(const float* __restrict__ W,
                                              const float* __restrict__ inv_norm,
                                              unsigned short* __restrict__ Wbf) {
    const size_t base = ((size_t)blockIdx.x * blockDim.x + threadIdx.x) * 8;
    const float s = inv_norm[base >> 11];  // D_ACT = 2048; 8 elems stay in one row
    float4 a = *reinterpret_cast<const float4*>(W + base);
    float4 b = *reinterpret_cast<const float4*>(W + base + 4);
    uint4 o;
    o.x = pack_bf16x2(a.x * s, a.y * s);
    o.y = pack_bf16x2(a.z * s, a.w * s);
    o.z = pack_bf16x2(b.x * s, b.y * s);
    o.w = pack_bf16x2(b.z * s, b.w * s);
    *reinterpret_cast<uint4*>(Wbf + base) = o;
}

// ---------- 3) X -> bf16 ----------
__global__ __launch_bounds__(256) void conv_x(const float* __restrict__ X,
                                              unsigned short* __restrict__ Xbf) {
    const size_t base = ((size_t)blockIdx.x * blockDim.x + threadIdx.x) * 8;
    float4 a = *reinterpret_cast<const float4*>(X + base);
    float4 b = *reinterpret_cast<const float4*>(X + base + 4);
    uint4 o;
    o.x = pack_bf16x2(a.x, a.y);
    o.y = pack_bf16x2(a.z, a.w);
    o.z = pack_bf16x2(b.x, b.y);
    o.w = pack_bf16x2(b.z, b.w);
    *reinterpret_cast<uint4*>(Xbf + base) = o;
}

// ---------- 4) scores = Xbf @ Wbf^T via WMMA bf16 ----------
// block = 128 threads (4 waves). Block tile: M=64 (16/wave), N=128, K-step 32.
// B tile (32K x 128N bf16 = 8 KB) staged in LDS once per block, double-buffered,
// via GLOBAL_LOAD_ASYNC_TO_LDS_B128 (ASYNCcnt) when available; waves read
// fragments with ds_load_b128. A fragments are direct global b128 loads.
#if defined(__HIP_DEVICE_COMPILE__) && \
    __has_builtin(__builtin_amdgcn_global_load_async_to_lds_b128)
#define GEMM_ASYNC 1
#else
#define GEMM_ASYNC 0
#endif

#if GEMM_ASYNC
#if __has_builtin(__builtin_amdgcn_s_wait_asynccnt)
#define WAIT_ASYNC0() __builtin_amdgcn_s_wait_asynccnt(0)
#else
#define WAIT_ASYNC0() asm volatile("s_wait_asynccnt 0" ::: "memory")
#endif
typedef int vec4i __attribute__((vector_size(16)));
typedef vec4i __attribute__((address_space(1)))* gvec_p;   // global (AS1)
typedef vec4i __attribute__((address_space(3)))* lvec_p;   // LDS (AS3)
__device__ __forceinline__ void async_cp16(const void* g, void* l) {
    __builtin_amdgcn_global_load_async_to_lds_b128(
        (gvec_p)(unsigned long long)g,
        (lvec_p)(unsigned int)(unsigned long long)l,
        /*offset=*/0, /*cpol=*/0);
}
#endif

__global__ __launch_bounds__(128) void gemm_wmma(const unsigned short* __restrict__ Xbf,
                                                 const unsigned short* __restrict__ Wbf,
                                                 float* __restrict__ out) {
    const int tid   = threadIdx.x;
    const int wave  = tid >> 5;
    const int lane  = tid & 31;
    const int lmod  = lane & 15;
    const int lhi   = lane >> 4;
    const int mBase = (blockIdx.y * 4 + wave) * 16;
    const int nBase = blockIdx.x * 128;

    // A frag (16x32 MxK, contiguous K): lanes 0-15 K {0..15}; lanes 16-31 K {16..31}
    const unsigned short* aBase = Xbf + (size_t)(mBase + lmod) * D_ACT + lhi * 16;

    v8f acc[8];
#pragma unroll
    for (int j = 0; j < 8; ++j) acc[j] = (v8f){0.f, 0.f, 0.f, 0.f, 0.f, 0.f, 0.f, 0.f};

#if GEMM_ASYNC
    __shared__ unsigned short Btile[2][128][32];   // [buf][n][k] : 16 KB
    const unsigned short* gB = Wbf + (size_t)(nBase + tid) * D_ACT;  // this thread's N-row
    // preload k=0 slice into buffer 0: 4 x b128 per thread
#pragma unroll
    for (int i = 0; i < 4; ++i) async_cp16(gB + i * 8, &Btile[0][tid][i * 8]);

    const int S = D_ACT / 32;
#pragma unroll 1
    for (int s = 0; s < S; ++s) {
        WAIT_ASYNC0();
        __syncthreads();                    // buffer (s&1) fully landed, visible to all
        if (s + 1 < S) {
            const unsigned short* gsrc = gB + (s + 1) * 32;
            unsigned short* ldst = &Btile[(s + 1) & 1][tid][0];
#pragma unroll
            for (int i = 0; i < 4; ++i) async_cp16(gsrc + i * 8, ldst + i * 8);
        }
        const int k = s * 32;
        v8bf alo = *reinterpret_cast<const v8bf*>(aBase + k);
        v8bf ahi = *reinterpret_cast<const v8bf*>(aBase + k + 8);
        v16bf a = __builtin_shufflevector(alo, ahi,
                     0, 1, 2, 3, 4, 5, 6, 7, 8, 9, 10, 11, 12, 13, 14, 15);
        const unsigned short (*bt)[32] = Btile[s & 1];
#pragma unroll
        for (int j = 0; j < 8; ++j) {
            v16bf b = *reinterpret_cast<const v16bf*>(&bt[j * 16 + lmod][lhi * 16]);
            acc[j] = __builtin_amdgcn_wmma_f32_16x16x32_bf16(
                         false, a, false, b, (short)0, acc[j], false, false);
        }
        __syncthreads();                    // all reads of buffer (s&1) done before reuse
    }
#else
    // fallback: direct global B fragments
    const unsigned short* bBase = Wbf + (size_t)(nBase + lmod) * D_ACT + lhi * 16;
#pragma unroll 2
    for (int k = 0; k < D_ACT; k += 32) {
        v8bf alo = *reinterpret_cast<const v8bf*>(aBase + k);
        v8bf ahi = *reinterpret_cast<const v8bf*>(aBase + k + 8);
        v16bf a = __builtin_shufflevector(alo, ahi,
                     0, 1, 2, 3, 4, 5, 6, 7, 8, 9, 10, 11, 12, 13, 14, 15);
#pragma unroll
        for (int j = 0; j < 8; ++j) {
            v16bf b = *reinterpret_cast<const v16bf*>(bBase + (size_t)j * 16 * D_ACT + k);
            acc[j] = __builtin_amdgcn_wmma_f32_16x16x32_bf16(
                         false, a, false, b, (short)0, acc[j], false, false);
        }
    }
#endif

    // C/D layout: VGPR r, lanes 0-15 -> M=r, N=lane; lanes 16-31 -> M=8+r, N=lane-16
    const int mOff = lhi * 8;
    const int nCol = nBase + lmod;
#pragma unroll
    for (int j = 0; j < 8; ++j) {
#pragma unroll
        for (int r = 0; r < 8; ++r) {
            out[(size_t)(mBase + mOff + r) * N_FEAT + nCol + j * 16] = acc[j][r];
        }
    }
}

// ---------- 5) per-row exact top-64 via 8-bit radix select, then relu, in place ----------
__global__ __launch_bounds__(256) void topk_kernel(float* __restrict__ out) {
    __shared__ unsigned int hist[256];
    __shared__ unsigned int s_prefix;
    __shared__ unsigned int s_needed;

    float* rowPtr = out + (size_t)blockIdx.x * N_FEAT;
    if (threadIdx.x == 0) { s_prefix = 0u; s_needed = SPARSITY; }

    for (int pass = 0; pass < 4; ++pass) {
        const int shift = 24 - 8 * pass;
        hist[threadIdx.x] = 0u;
        __syncthreads();
        const unsigned int pfx = s_prefix;  // bits above (shift+8)
        for (int i = threadIdx.x; i < N_FEAT; i += 256) {
            const unsigned int key = float_key(rowPtr[i]);
            if (pass == 0 || (key >> (shift + 8)) == pfx)
                atomicAdd(&hist[(key >> shift) & 255u], 1u);
        }
        __syncthreads();
        if (threadIdx.x == 0) {
            unsigned int need = s_needed, cum = 0;
            int b = 255;
            for (; b > 0; --b) {
                const unsigned int c = hist[b];
                if (cum + c >= need) break;
                cum += c;
            }
            s_prefix = (s_prefix << 8) | (unsigned int)b;
            s_needed = need - cum;
        }
        __syncthreads();
    }

    const unsigned int thrKey = s_prefix;  // exact key of the 64th-largest value
    for (int i = threadIdx.x; i < N_FEAT; i += 256) {
        const float v = rowPtr[i];
        const float o = (float_key(v) >= thrKey) ? fmaxf(v, 0.0f) : 0.0f;
        rowPtr[i] = o;
    }
}

// ---------- launcher ----------
extern "C" void kernel_launch(void* const* d_in, const int* in_sizes, int n_in,
                              void* d_out, int out_size, void* d_ws, size_t ws_size,
                              hipStream_t stream) {
    const float* x = (const float*)d_in[0];      // [BATCH, D_ACT]
    const float* w = (const float*)d_in[1];      // [N_FEAT, D_ACT]
    float* out = (float*)d_out;                  // [BATCH, N_FEAT]

    float* inv_norm = (float*)d_ws;                                   // 128 KB
    unsigned short* Xbf = (unsigned short*)((char*)d_ws + (size_t)N_FEAT * 4);
    unsigned short* Wbf = (unsigned short*)((char*)d_ws + (size_t)N_FEAT * 4
                                            + (size_t)BATCH * D_ACT * 2);

    norm_kernel<<<N_FEAT / 8, 256, 0, stream>>>(w, inv_norm);
    conv_w<<<(N_FEAT * (size_t)D_ACT) / (256 * 8), 256, 0, stream>>>(w, inv_norm, Wbf);
    conv_x<<<(BATCH * (size_t)D_ACT) / (256 * 8), 256, 0, stream>>>(x, Xbf);
    gemm_wmma<<<dim3(N_FEAT / 128, BATCH / 64), 128, 0, stream>>>(Xbf, Wbf, out);
    topk_kernel<<<BATCH, 256, 0, stream>>>(out);
}
